// RNetvH_61349312856334
// MI455X (gfx1250) — matrix operand order repaired and checked by reading, not compiled, not run on hardware
//
#include <hip/hip_runtime.h>
#include <stdint.h>

// Problem constants (match reference)
#define BB 128
#define TT 24
#define EE 512
#define HH 512
#define VV 10000
#define GG (4 * HH)   // 2048

typedef __attribute__((ext_vector_type(16))) __bf16        bf16x16;
typedef __attribute__((ext_vector_type(8)))  float         f32x8;
typedef __attribute__((ext_vector_type(4)))  unsigned int  u32x4;

// ---------------------------------------------------------------- helpers ---

__device__ __forceinline__ uint16_t f2bf(float f) {
  union { float f; uint32_t u; } c; c.f = f;
  uint32_t u = c.u;
  // round-to-nearest-even f32 -> bf16
  uint32_t r = (u + 0x7FFFu + ((u >> 16) & 1u)) >> 16;
  return (uint16_t)r;
}

__device__ __forceinline__ float sigmoidf(float x) {
  return 1.0f / (1.0f + __expf(-x));
}

// Assemble a 16-element bf16 fragment from two 16-byte contiguous runs.
__device__ __forceinline__ bf16x16 ld_frag(const uint16_t* __restrict__ p0,
                                           const uint16_t* __restrict__ p1) {
  union { bf16x16 f; u32x4 u[2]; } r;
  r.u[0] = *(const u32x4*)p0;
  r.u[1] = *(const u32x4*)p1;
  return r.f;
}

// ---------------------------------------------------------------- kernels ---

__global__ void k_f32_to_bf16(const float* __restrict__ in,
                              uint16_t* __restrict__ out, int n) {
  int i = blockIdx.x * blockDim.x + threadIdx.x;
  if (i < n) out[i] = f2bf(in[i]);
}

__global__ void k_bias_sum(const float* __restrict__ a,
                           const float* __restrict__ b,
                           float* __restrict__ o) {
  int i = blockIdx.x * blockDim.x + threadIdx.x;
  if (i < GG) o[i] = a[i] + b[i];
}

__global__ void k_embed(const int* __restrict__ caps,
                        const float* __restrict__ Wemb,
                        uint16_t* __restrict__ xbf) {
  int i = blockIdx.x * blockDim.x + threadIdx.x;
  if (i >= BB * TT * EE) return;
  int bt = i / EE;
  int e  = i - bt * EE;
  int tok = caps[bt];
  float v = (tok == 0) ? 0.0f : Wemb[(size_t)tok * EE + e];  // padding_idx = 0
  xbf[i] = f2bf(v);
}

__global__ void k_init_state(const float* __restrict__ images,
                             uint16_t* __restrict__ hbf,
                             float* __restrict__ c) {
  int i = blockIdx.x * blockDim.x + threadIdx.x;
  if (i >= BB * HH) return;
  hbf[i] = f2bf(images[i]);   // h0 = images
  c[i]   = 0.0f;              // c0 = 0
}

__global__ void k_start(float* __restrict__ out) {
  int i = blockIdx.x * blockDim.x + threadIdx.x;
  if (i >= BB * VV) return;
  int b = i / VV;
  int j = i - b * VV;
  out[(size_t)b * (TT + 1) * VV + j] = (j == 1) ? 1.0f : 0.0f;  // one-hot <SOS>
}

__global__ void k_lengths(const int* __restrict__ len, int* __restrict__ out) {
  int i = threadIdx.x;
  if (i < BB) out[i] = len[i] - 1;
}

__global__ void k_lstm_cell(const float* __restrict__ gates,
                            float* __restrict__ c,
                            uint16_t* __restrict__ hbf) {
  int i = blockIdx.x * blockDim.x + threadIdx.x;
  if (i >= BB * HH) return;
  int b = i / HH;
  int j = i - b * HH;
  const float* g = gates + (size_t)b * GG;
  float gi = sigmoidf(g[j]);
  float gf = sigmoidf(g[HH + j]);
  float gg = tanhf(g[2 * HH + j]);
  float go = sigmoidf(g[3 * HH + j]);
  float cn = gf * c[i] + gi * gg;
  c[i] = cn;
  hbf[i] = f2bf(go * tanhf(cn));
}

// One K-pass of the WMMA GEMM: acc += A[m0..m0+31, 0..K) * W[ncol, 0..K)^T.
// Single-buffered fragments (no spills; ~112 live VGPRs). Per k-chunk: one
// clause of 12 global_load_b128, one wait, 8 back-to-back WMMAs. Load latency
// is hidden by multi-wave occupancy (1-wave workgroups, many blocks in flight),
// not by intra-wave pipelining.
__device__ __forceinline__ void
gemm_pass(f32x8 (&acc)[2][4],
          const uint16_t* __restrict__ A, int lda,
          const uint16_t* __restrict__ W, int ldw, int K,
          int m0, const int (&ncol)[4], int hf, int ln) {
  // A fragment (16x32 bf16): lane holds row m0+ln, K runs {hf*8..+7, 16+hf*8..+7}
  const uint16_t* pa0 = A + (size_t)(m0 + ln) * lda + hf * 8;
  const uint16_t* pa1 = pa0 + (size_t)16 * lda;
  // B fragment (32x16 bf16 of W^T): lane holds col ncol+ln, K runs {hf*8..+7, 16+hf*8..+7}
  const uint16_t* pb0 = W + (size_t)(ncol[0] + ln) * ldw + hf * 8;
  const uint16_t* pb1 = W + (size_t)(ncol[1] + ln) * ldw + hf * 8;
  const uint16_t* pb2 = W + (size_t)(ncol[2] + ln) * ldw + hf * 8;
  const uint16_t* pb3 = W + (size_t)(ncol[3] + ln) * ldw + hf * 8;

  for (int k = 0; k < K; k += 32) {
    bf16x16 a0 = ld_frag(pa0, pa0 + 16);
    bf16x16 a1 = ld_frag(pa1, pa1 + 16);
    bf16x16 b0 = ld_frag(pb0, pb0 + 16);
    bf16x16 b1 = ld_frag(pb1, pb1 + 16);
    bf16x16 b2 = ld_frag(pb2, pb2 + 16);
    bf16x16 b3 = ld_frag(pb3, pb3 + 16);
    acc[0][0] = __builtin_amdgcn_wmma_f32_16x16x32_bf16(false, a0, false, b0, (short)0, acc[0][0], false, false);
    acc[1][0] = __builtin_amdgcn_wmma_f32_16x16x32_bf16(false, a1, false, b0, (short)0, acc[1][0], false, false);
    acc[0][1] = __builtin_amdgcn_wmma_f32_16x16x32_bf16(false, a0, false, b1, (short)0, acc[0][1], false, false);
    acc[1][1] = __builtin_amdgcn_wmma_f32_16x16x32_bf16(false, a1, false, b1, (short)0, acc[1][1], false, false);
    acc[0][2] = __builtin_amdgcn_wmma_f32_16x16x32_bf16(false, a0, false, b2, (short)0, acc[0][2], false, false);
    acc[1][2] = __builtin_amdgcn_wmma_f32_16x16x32_bf16(false, a1, false, b2, (short)0, acc[1][2], false, false);
    acc[0][3] = __builtin_amdgcn_wmma_f32_16x16x32_bf16(false, a0, false, b3, (short)0, acc[0][3], false, false);
    acc[1][3] = __builtin_amdgcn_wmma_f32_16x16x32_bf16(false, a1, false, b3, (short)0, acc[1][3], false, false);
    pa0 += 32; pa1 += 32; pb0 += 32; pb1 += 32; pb2 += 32; pb3 += 32;
  }
}

// C[M x N] = A[M x K] * W[N x K]^T (+ optional A2 * W2^T) + bias.
// One wave per 32x64 output block; bf16 WMMA, fp32 accumulate.
// M multiple of 32; N multiple of 16. Edge blocks clamp their n-tiles onto the
// last valid 16-column tile (branch-free; duplicated tiles store identical
// values), so EXEC stays all-ones and the inner loop has no divergence.
__global__ void k_gemm_bf16(const uint16_t* __restrict__ A,  int lda,
                            const uint16_t* __restrict__ W,
                            const uint16_t* __restrict__ A2, int lda2,
                            const uint16_t* __restrict__ W2,
                            int K, int K2,
                            float* __restrict__ C, long long ldc,
                            const float* __restrict__ bias,
                            int N) {
  const int lane = threadIdx.x & 31;
  const int hf = lane >> 4;
  const int ln = lane & 15;
  const int m0 = blockIdx.y * 32;
  const int n0 = blockIdx.x * 64;

  int ncol[4];
#pragma unroll
  for (int nt = 0; nt < 4; ++nt) {
    int cc = n0 + nt * 16;
    ncol[nt] = (cc > N - 16) ? (N - 16) : cc;  // wave-uniform clamp
  }

  const f32x8 z = {0.f, 0.f, 0.f, 0.f, 0.f, 0.f, 0.f, 0.f};
  f32x8 acc[2][4];
#pragma unroll
  for (int mt = 0; mt < 2; ++mt)
#pragma unroll
    for (int nt = 0; nt < 4; ++nt) acc[mt][nt] = z;

  gemm_pass(acc, A, lda, W, K, K, m0, ncol, hf, ln);
  if (A2 != nullptr)  // fused second operand pair (h * W_hh^T)
    gemm_pass(acc, A2, lda2, W2, K2, K2, m0, ncol, hf, ln);

  // Epilogue: C/D layout -> lane covers col ncol[nt]+ln,
  // VGPR r covers row m0 + mt*16 + hf*8 + r
#pragma unroll
  for (int mt = 0; mt < 2; ++mt) {
#pragma unroll
    for (int nt = 0; nt < 4; ++nt) {
      const int n = ncol[nt] + ln;
      const float bv = bias ? bias[n] : 0.0f;
      float* pc = C + (size_t)(m0 + mt * 16 + hf * 8) * ldc + n;
#pragma unroll
      for (int r = 0; r < 8; ++r) {
        *pc = acc[mt][nt][r] + bv;
        pc += ldc;
      }
    }
  }
}

// ------------------------------------------------------------------ host ---

extern "C" void kernel_launch(void* const* d_in, const int* in_sizes, int n_in,
                              void* d_out, int out_size, void* d_ws, size_t ws_size,
                              hipStream_t stream) {
  (void)in_sizes; (void)n_in; (void)out_size; (void)ws_size;

  const float* images   = (const float*)d_in[0];
  const int*   captions = (const int*)  d_in[1];
  const int*   cap_len  = (const int*)  d_in[2];
  const float* W_emb    = (const float*)d_in[3];
  const float* W_ih     = (const float*)d_in[4];
  const float* W_hh     = (const float*)d_in[5];
  const float* b_ih     = (const float*)d_in[6];
  const float* b_hh     = (const float*)d_in[7];
  const float* W_out    = (const float*)d_in[8];
  const float* b_out    = (const float*)d_in[9];
  float* out = (float*)d_out;

  // Workspace carve-up (256B aligned slices)
  uint8_t* ws = (uint8_t*)d_ws;
  size_t off = 0;
  auto alloc = [&](size_t bytes) -> uint8_t* {
    uint8_t* p = ws + off;
    off += (bytes + 255) & ~(size_t)255;
    return p;
  };
  uint16_t* Wih_bf   = (uint16_t*)alloc((size_t)GG * EE * 2);
  uint16_t* Whh_bf   = (uint16_t*)alloc((size_t)GG * HH * 2);
  uint16_t* Wout_bf  = (uint16_t*)alloc((size_t)VV * HH * 2);
  uint16_t* x_bf     = (uint16_t*)alloc((size_t)BB * TT * EE * 2);
  float*    bias_sum = (float*)   alloc((size_t)GG * 4);
  uint16_t* h_bf     = (uint16_t*)alloc((size_t)BB * HH * 2);
  float*    cbuf     = (float*)   alloc((size_t)BB * HH * 4);
  float*    gates    = (float*)   alloc((size_t)BB * GG * 4);

  // Prologue: weight/activation precision conversion + state init + t=0 output
  k_f32_to_bf16<<<(GG * EE + 255) / 256, 256, 0, stream>>>(W_ih, Wih_bf, GG * EE);
  k_f32_to_bf16<<<(GG * HH + 255) / 256, 256, 0, stream>>>(W_hh, Whh_bf, GG * HH);
  k_f32_to_bf16<<<(VV * HH + 255) / 256, 256, 0, stream>>>(W_out, Wout_bf, VV * HH);
  k_bias_sum<<<(GG + 255) / 256, 256, 0, stream>>>(b_ih, b_hh, bias_sum);
  k_embed<<<(BB * TT * EE + 255) / 256, 256, 0, stream>>>(captions, W_emb, x_bf);
  k_init_state<<<(BB * HH + 255) / 256, 256, 0, stream>>>(images, h_bf, cbuf);
  k_start<<<(BB * VV + 255) / 256, 256, 0, stream>>>(out);
  k_lengths<<<1, 128, 0, stream>>>(cap_len, (int*)(out + (size_t)BB * (TT + 1) * VV));

  const dim3 gGates(GG / 64, BB / 32);            // 32 x 4 waves
  const dim3 gLogits((VV + 63) / 64, BB / 32);    // 157 x 4 waves
  for (int t = 0; t < TT; ++t) {
    // gates = x_t * W_ih^T + h * W_hh^T + (b_ih + b_hh)   [fused dual-GEMM]
    k_gemm_bf16<<<gGates, 32, 0, stream>>>(
        x_bf + (size_t)t * EE, TT * EE, Wih_bf,
        h_bf, HH, Whh_bf,
        EE, HH,
        gates, (long long)GG,
        bias_sum, GG);
    // c,h update (elementwise)
    k_lstm_cell<<<(BB * HH + 255) / 256, 256, 0, stream>>>(gates, cbuf, h_bf);
    // out[:, t+1, :] = h * W_out^T + b_out
    k_gemm_bf16<<<gLogits, 32, 0, stream>>>(
        h_bf, HH, Wout_bf,
        (const uint16_t*)nullptr, 0, (const uint16_t*)nullptr,
        HH, 0,
        out + (size_t)(t + 1) * VV, (long long)(TT + 1) * VV,
        b_out, VV);
  }
}
